// AttentionHead_42202348650661
// MI455X (gfx1250) — matrix operand order repaired and checked by reading, not compiled, not run on hardware
//
#include <hip/hip_runtime.h>
#include <hip/hip_bf16.h>

typedef __attribute__((ext_vector_type(16))) __bf16 v16bf;
typedef __attribute__((ext_vector_type(8)))  __bf16 v8bf;
typedef __attribute__((ext_vector_type(4)))  __bf16 v4bf;
typedef __attribute__((ext_vector_type(8)))  float  v8f;

#define NB 4
#define NS 2048
#define NE 1024
#define ND 64

#define NEG_INF (-__builtin_inff())

static __device__ __forceinline__ v8f zero8() {
  v8f z;
#pragma unroll
  for (int i = 0; i < 8; ++i) z[i] = 0.0f;
  return z;
}

// A-fragment (16x32 bf16, wave32): lanes 0-15 (M=lane): elems 0..15 -> K=0..15;
// lanes 16-31 (M=lane-16): elems 0..15 -> K=16..31.
static __device__ __forceinline__ v16bf load_a_frag(const __bf16* rowp, int hi) {
  const v8bf* p0 = (const v8bf*)(rowp + hi * 16);
  const v8bf* p1 = (const v8bf*)(rowp + hi * 16 + 8);
  v8bf lo = *p0;
  v8bf hi8 = *p1;
  v16bf r;
#pragma unroll
  for (int i = 0; i < 8; ++i) { r[i] = lo[i]; r[i + 8] = hi8[i]; }
  return r;
}

static __device__ __forceinline__ v8f wmma_bf16(v16bf a, v16bf b, v8f c) {
  return __builtin_amdgcn_wmma_f32_16x16x32_bf16(false, a, false, b, (short)0, c,
                                                 false, false);
}

static __device__ __forceinline__ float red_max16(float v) {
  v = fmaxf(v, __shfl_xor(v, 1, 32));
  v = fmaxf(v, __shfl_xor(v, 2, 32));
  v = fmaxf(v, __shfl_xor(v, 4, 32));
  v = fmaxf(v, __shfl_xor(v, 8, 32));
  return v;
}

static __device__ __forceinline__ float red_sum16(float v) {
  v += __shfl_xor(v, 1, 32);
  v += __shfl_xor(v, 2, 32);
  v += __shfl_xor(v, 4, 32);
  v += __shfl_xor(v, 8, 32);
  return v;
}

// ---------------- pre-pass: fp32 -> bf16 conversions / packing ----------------

__global__ void cvt_x_kernel(const float* __restrict__ x, __bf16* __restrict__ xbf) {
  const size_t i = ((size_t)blockIdx.x * blockDim.x + threadIdx.x) * 4;
  const float4 f = *(const float4*)(x + i);
  v4bf o;
  o[0] = (__bf16)f.x; o[1] = (__bf16)f.y; o[2] = (__bf16)f.z; o[3] = (__bf16)f.w;
  *(v4bf*)(xbf + i) = o;
}

// Wpack layout per projection: idx = ((kt*64) + n)*32 + kk  (kt = K/32, kk = K%32)
__global__ void pack_w_kernel(const float* __restrict__ Wq,
                              const float* __restrict__ Wk,
                              const float* __restrict__ Wv,
                              __bf16* __restrict__ wpack) {
  const int i = blockIdx.x * blockDim.x + threadIdx.x;
  if (i >= 3 * NE * ND) return;
  const int proj = i / (NE * ND);
  const int r = i % (NE * ND);
  const int kt = r / (ND * 32);
  const int rem = r % (ND * 32);
  const int n = rem / 32;
  const int kk = rem % 32;
  const float* W = (proj == 0) ? Wq : (proj == 1) ? Wk : Wv;
  wpack[i] = (__bf16)W[(size_t)(kt * 32 + kk) * ND + n];
}

// ---------------- QKV projection -----------------------------------------------
// 16 rows x 64 cols per wave (4 accum tiles).  kt-loop 2x-stepped with
// ping-pong fragment buffers (no rotate copies).  Q pre-scaled by 1/64
// (exact power of two).  q/k row-major bf16, v transposed [B][D][S].

__global__ void __launch_bounds__(128)
qkv_proj_kernel(const __bf16* __restrict__ xbf,
                const __bf16* __restrict__ wpack,
                const float* __restrict__ bq,
                const float* __restrict__ bk,
                const float* __restrict__ bv,
                __bf16* __restrict__ qbf,
                __bf16* __restrict__ kbf,
                __bf16* __restrict__ vT) {
  const int lane = threadIdx.x & 31;
  const int wave = threadIdx.x >> 5;
  const int hi = lane >> 4;
  const int ln = lane & 15;
  const int tile = blockIdx.x * 4 + wave;   // 16-row tile of [B*S]
  const int proj = blockIdx.y;
  const int row0 = tile * 16;

  const __bf16* wp = wpack + (size_t)proj * (NE * ND);
  const float* bias = (proj == 0) ? bq : (proj == 1) ? bk : bv;

  v8f acc[4];
#pragma unroll
  for (int nt = 0; nt < 4; ++nt) acc[nt] = zero8();

  const __bf16* arow = xbf + (size_t)(row0 + ln) * NE;

  auto load_bfrags = [&](int kt, v16bf* bf) {
#pragma unroll
    for (int nt = 0; nt < 4; ++nt)
      bf[nt] = *(const v16bf*)(wp + (size_t)(kt * 64 + nt * 16 + ln) * 32 + hi * 16);
  };

  // ping-pong pipeline over NE/32 = 32 k-steps (even steps -> A, odd -> B)
  v16bf aA, aB, bA[4], bB[4];
  aA = load_a_frag(arow, hi);
  load_bfrags(0, bA);

  for (int k2 = 0; k2 < NE / 64; ++k2) {
    const int kt1 = 2 * k2 + 1;
    aB = load_a_frag(arow + kt1 * 32, hi);          // odd-step fragments
    load_bfrags(kt1, bB);

#pragma unroll
    for (int nt = 0; nt < 4; ++nt) acc[nt] = wmma_bf16(aA, bA[nt], acc[nt]);

    if (k2 < NE / 64 - 1) {                         // even-step fragments
      const int kt2 = 2 * k2 + 2;
      aA = load_a_frag(arow + kt2 * 32, hi);
      load_bfrags(kt2, bA);
    }

#pragma unroll
    for (int nt = 0; nt < 4; ++nt) acc[nt] = wmma_bf16(aB, bB[nt], acc[nt]);
  }

  const float qscale = (proj == 0) ? (1.0f / (float)ND) : 1.0f;
#pragma unroll
  for (int nt = 0; nt < 4; ++nt) {
    const int col = nt * 16 + ln;
    const float bb = bias[col];
#pragma unroll
    for (int r = 0; r < 8; ++r) {
      const int srow = row0 + r + 8 * hi;   // global row in [B*S]
      const float val = (acc[nt][r] + bb) * qscale;
      if (proj == 0) {
        qbf[(size_t)srow * ND + col] = (__bf16)val;
      } else if (proj == 1) {
        kbf[(size_t)srow * ND + col] = (__bf16)val;
      } else {
        const int bidx = srow / NS;
        const int s = srow % NS;
        vT[((size_t)bidx * ND + col) * NS + s] = (__bf16)val;
      }
    }
  }
}

// ---------------- flash attention (causal, online softmax) -------------------
// Key-block loop 2x-stepped with ping-pong K-fragment buffers (no copies).
// V fragments load at block top; consumed only after the softmax.

__global__ void __launch_bounds__(128)
attn_kernel(const __bf16* __restrict__ qbf,
            const __bf16* __restrict__ kbf,
            const __bf16* __restrict__ vT,
            float* __restrict__ out) {
  __shared__ __align__(64) __bf16 plds_all[4][16 * 32];
  const int lane = threadIdx.x & 31;
  const int wave = threadIdx.x >> 5;
  const int hi = lane >> 4;
  const int ln = lane & 15;
  const int t = blockIdx.x * 4 + wave;      // 16-query tile within batch
  const int b = blockIdx.y;
  const int q0 = t * 16;
  __bf16* plds = plds_all[wave];

  const __bf16* qb = qbf + (size_t)b * NS * ND;
  const __bf16* kb = kbf + (size_t)b * NS * ND;
  const __bf16* vb = vT + (size_t)b * ND * NS;

  // Q fragments are loop-invariant (already scaled by 1/64 at projection).
  const v16bf aq0 = load_a_frag(qb + (size_t)(q0 + ln) * ND + 0, hi);
  const v16bf aq1 = load_a_frag(qb + (size_t)(q0 + ln) * ND + 32, hi);

  float m_i[8], l_i[8];
  v8f o[4];
#pragma unroll
  for (int r = 0; r < 8; ++r) { m_i[r] = NEG_INF; l_i[r] = 0.0f; }
#pragma unroll
  for (int dt = 0; dt < 4; ++dt) o[dt] = zero8();

  const int nkb = (q0 + 15) / 32 + 1;       // key blocks of 32 up to diagonal

  // kf[dstep*2 + subtile]: B-fragments of K for one 32-key block
  auto load_kfrags = [&](int kb0, v16bf* kf) {
#pragma unroll
    for (int dstep = 0; dstep < 2; ++dstep) {
      kf[dstep * 2 + 0] = *(const v16bf*)(kb + (size_t)(kb0 + ln) * ND + dstep * 32 + hi * 16);
      kf[dstep * 2 + 1] = *(const v16bf*)(kb + (size_t)(kb0 + 16 + ln) * ND + dstep * 32 + hi * 16);
    }
  };

  auto process = [&](int kblk, const v16bf* kf) {
    const int kb0 = kblk * 32;

    // V fragments issued now, consumed only after the softmax below.
    v16bf vf[4];
#pragma unroll
    for (int dt = 0; dt < 4; ++dt)
      vf[dt] = *(const v16bf*)(vb + (size_t)(dt * 16 + ln) * NS + kb0 + hi * 16);

    v8f s0 = zero8(), s1 = zero8();
    s0 = wmma_bf16(aq0, kf[0], s0);
    s1 = wmma_bf16(aq0, kf[1], s1);
    s0 = wmma_bf16(aq1, kf[2], s0);
    s1 = wmma_bf16(aq1, kf[3], s1);

    if (kb0 + 2 * 32 - 1 < (q0 + 15) + 32) {        // prefetch 2 blocks ahead
      __builtin_prefetch(kb + (size_t)(kb0 + 64 + lane) * ND, 0, 2);
      __builtin_prefetch(vb + (size_t)(lane * 2) * NS + kb0 + 64, 0, 2);
      __builtin_prefetch(vb + (size_t)(lane * 2 + 1) * NS + kb0 + 64, 0, 2);
    }

    if (kb0 + 31 > q0) {                    // block may contain masked entries
#pragma unroll
      for (int r = 0; r < 8; ++r) {
        const int qrow = q0 + r + 8 * hi;
        if (kb0 + ln > qrow) s0[r] = NEG_INF;
        if (kb0 + 16 + ln > qrow) s1[r] = NEG_INF;
      }
    }

    // online softmax; each row lives across 16 lanes of one half-wave
#pragma unroll
    for (int r = 0; r < 8; ++r) {
      float v = fmaxf(s0[r], s1[r]);
      v = red_max16(v);
      const float mnew = fmaxf(m_i[r], v);
      const float alpha = __expf(m_i[r] - mnew);
      const float p0 = __expf(s0[r] - mnew);
      const float p1 = __expf(s1[r] - mnew);
      const float rs = red_sum16(p0 + p1);
      l_i[r] = l_i[r] * alpha + rs;
      m_i[r] = mnew;
      const int m = r + 8 * hi;
      plds[m * 32 + ln] = (__bf16)p0;       // C-layout -> row-major P in LDS
      plds[m * 32 + 16 + ln] = (__bf16)p1;
#pragma unroll
      for (int dt = 0; dt < 4; ++dt) o[dt][r] *= alpha;
    }

    asm volatile("s_wait_dscnt 0x0" ::: "memory");   // LDS stores visible (same wave)

    const v16bf ap = load_a_frag(&plds[ln * 32], hi);
#pragma unroll
    for (int dt = 0; dt < 4; ++dt) o[dt] = wmma_bf16(ap, vf[dt], o[dt]);
  };

  // ping-pong over key blocks: even blocks use kfA, odd blocks use kfB
  v16bf kfA[4], kfB[4];
  load_kfrags(0, kfA);

  for (int base = 0; base < nkb; base += 2) {
    if (base + 1 < nkb) load_kfrags((base + 1) * 32, kfB);
    process(base, kfA);
    if (base + 1 < nkb) {
      if (base + 2 < nkb) load_kfrags((base + 2) * 32, kfA);
      process(base + 1, kfB);
    }
  }

  // epilogue: normalize and store fp32
#pragma unroll
  for (int r = 0; r < 8; ++r) {
    const float inv = 1.0f / l_i[r];
    const int srow = q0 + r + 8 * hi;
#pragma unroll
    for (int dt = 0; dt < 4; ++dt) {
      out[((size_t)b * NS + srow) * ND + dt * 16 + ln] = o[dt][r] * inv;
    }
  }
}

// ---------------- host launch -------------------------------------------------

extern "C" void kernel_launch(void* const* d_in, const int* in_sizes, int n_in,
                              void* d_out, int out_size, void* d_ws, size_t ws_size,
                              hipStream_t stream) {
  (void)in_sizes; (void)n_in; (void)out_size; (void)ws_size;
  const float* x  = (const float*)d_in[0];
  const float* Wq = (const float*)d_in[1];
  const float* bq = (const float*)d_in[2];
  const float* Wk = (const float*)d_in[3];
  const float* bk = (const float*)d_in[4];
  const float* Wv = (const float*)d_in[5];
  const float* bv = (const float*)d_in[6];
  float* out = (float*)d_out;

  __bf16* xbf   = (__bf16*)d_ws;                         // B*S*E
  __bf16* wpack = xbf + (size_t)NB * NS * NE;            // 3*E*D
  __bf16* qbf   = wpack + (size_t)3 * NE * ND;           // B*S*D (pre-scaled 1/64)
  __bf16* kbf   = qbf + (size_t)NB * NS * ND;            // B*S*D
  __bf16* vT    = kbf + (size_t)NB * NS * ND;            // B*D*S (transposed)

  const int n_x = NB * NS * NE;
  cvt_x_kernel<<<(n_x / 4 + 255) / 256, 256, 0, stream>>>(x, xbf);
  pack_w_kernel<<<(3 * NE * ND + 255) / 256, 256, 0, stream>>>(Wq, Wk, Wv, wpack);

  dim3 gproj(NB * NS / 16 / 4, 3);
  qkv_proj_kernel<<<gproj, 128, 0, stream>>>(xbf, wpack, bq, bk, bv, qbf, kbf, vT);

  dim3 gattn(NS / 16 / 4, NB);
  attn_kernel<<<gattn, 128, 0, stream>>>(qbf, kbf, vT, out);
}
